// Model_71201967833825
// MI455X (gfx1250) — matrix operand order repaired and checked by reading, not compiled, not run on hardware
//
#include <hip/hip_runtime.h>
#include <hip/hip_bf16.h>

#define NB (8)     // B
#define NP (512)   // P
#define NQ (64)    // Q
#define NA (16)    // A
#define NH (128)   // H
#define NH2 (256)
#define NH10 (1280)
#define NG (384)
#define NEG_INF (-1e30f)

typedef __attribute__((ext_vector_type(16))) _Float16 v16h;
typedef __attribute__((ext_vector_type(8)))  float    v8f;

// ---------------- WMMA fragment helpers (CDNA5 16x16x32 f16) ----------------
// A-matrix 16x32 layout (matches packed B): lane<16 -> row=lane, K in {0..15};
// lane>=16 -> row=lane-16, K in {16..31}. Each lane's fragment is a
// contiguous 32-byte run -> compiler emits 2x (global|ds)_load_b128.
__device__ __forceinline__ v16h load_frag_a(const _Float16* A, int lda, int lane) {
  v16h f;
  const _Float16* row = A + (lane & 15) * lda;
  const int hi = (lane >= 16) ? 16 : 0;
#pragma unroll
  for (int v = 0; v < 8; ++v) {
    const int col = 2 * v + hi;
    f[2 * v]     = row[col];
    f[2 * v + 1] = row[col + 1];
  }
  return f;
}

// B fragments come from a pre-packed buffer: for k-tile kt (32 rows of B),
// n-tile nt (16 cols), lane l holds 16 contiguous halfs:
//   Bp[((kt*ntiles + nt)*32 + l)*16 + e] = B[kt*32 + (l>=16?16:0) + e][nt*16 + (l&15)]
__device__ __forceinline__ v16h load_frag_bp(const _Float16* Bp, int ktile,
                                             int ntile, int ntiles, int lane) {
  return *(const v16h*)(Bp + (((size_t)ktile * ntiles + ntile) * 32 + lane) * 16);
}

__device__ __forceinline__ v8f wmma32(v16h a, v16h b, v8f c) {
  return __builtin_amdgcn_wmma_f32_16x16x32_f16(false, a, false, b, (short)0, c,
                                                false, false);
}

// ---------------- f32 -> f16 convert (row-major A operands) ----------------
__global__ void k_cvt(const float* __restrict__ s, _Float16* __restrict__ d, int n) {
  int i = blockIdx.x * 256 + threadIdx.x;
  if (i < n) d[i] = (_Float16)s[i];
}

// ---------------- f32 -> f16 convert + pack into B-fragment layout ----------
__global__ void k_pack_b(const float* __restrict__ src, _Float16* __restrict__ dst,
                         int K, int N) {
  const int i = blockIdx.x * 256 + threadIdx.x;
  if (i >= K * N) return;
  const int e    = i & 15;          // element within lane chunk -> k offset
  const int l    = (i >> 4) & 31;   // lane
  const int tile = i >> 9;          // kt*(N/16) + nt
  const int ntiles = N >> 4;
  const int nt = tile % ntiles;
  const int kt = tile / ntiles;
  const int k = kt * 32 + ((l >= 16) ? 16 : 0) + e;
  const int n = nt * 16 + (l & 15);
  dst[i] = (_Float16)src[(size_t)k * N + n];
}

// ---- generic WMMA GEMM: C(MxN f32) = A(MxK f16 row-major) * Bp(packed) + bias
// Each wave holds one B fragment per K-step and accumulates four M-tiles
// (16x64-per-wave, 64x64 per 128-thread block): 2.5 b128 loads per v_wmma.
__global__ void __launch_bounds__(128) k_wmma_gemm(
    const _Float16* __restrict__ A, const _Float16* __restrict__ Bp,
    const float* __restrict__ bias, float* __restrict__ C, int M, int N, int K) {
  const int lane   = threadIdx.x & 31;
  const int wave   = threadIdx.x >> 5;
  const int ntiles = N >> 4;
  const int ntile  = blockIdx.x * 4 + wave;
  const int nbase  = ntile * 16;
  const int mbase  = blockIdx.y * 64;
  v8f acc[4] = {};
  for (int k0 = 0; k0 < K; k0 += 32) {
    v16h b = load_frag_bp(Bp, k0 >> 5, ntile, ntiles, lane);
#pragma unroll
    for (int j = 0; j < 4; ++j) {
      v16h a = load_frag_a(A + (size_t)(mbase + 16 * j) * K + k0, K, lane);
      acc[j] = wmma32(a, b, acc[j]);
    }
  }
  const int col  = nbase + (lane & 15);
  const float bv = bias ? bias[col] : 0.0f;
#pragma unroll
  for (int j = 0; j < 4; ++j) {
    const int rbase = mbase + 16 * j + ((lane >= 16) ? 8 : 0);
#pragma unroll
    for (int v = 0; v < 8; ++v)
      C[(size_t)(rbase + v) * N + col] = acc[j][v] + bv;
  }
}

// ---------------- alternative pooling (a1/a2/a3) ----------------
__global__ void __launch_bounds__(128) k_alt_pool(
    const float* __restrict__ x, const int* __restrict__ mask,
    const float* __restrict__ w, float* __restrict__ out, int ai) {
  const int b = blockIdx.x;
  __shared__ float sc[NA], al[NA];
  const float* xb = x + (size_t)b * NA * NH;
  const int t = threadIdx.x;
  if (t < NA) {
    float s = 0.f;
    for (int h = 0; h < NH; ++h) s += xb[t * NH + h] * w[h];
    sc[t] = mask[b * NA + t] ? s : NEG_INF;
  }
  __syncthreads();
  if (t == 0) {
    float m = NEG_INF;
    for (int i = 0; i < NA; ++i) m = fmaxf(m, sc[i]);
    float ss = 0.f;
    for (int i = 0; i < NA; ++i) { al[i] = __expf(sc[i] - m); ss += al[i]; }
    float inv = 1.f / ss;
    for (int i = 0; i < NA; ++i) al[i] *= inv;
  }
  __syncthreads();
  if (t < NH) {
    float r = 0.f;
    for (int i = 0; i < NA; ++i) r += al[i] * xb[i * NH + t];
    out[(b * 3 + ai) * NH + t] = r;
  }
}

// ---------------- fused attentions c, b, m (scores+softmax+attend) ----------
__global__ void __launch_bounds__(256) k_attn_cbm(
    const float* __restrict__ passage, const float* __restrict__ query,
    const float* __restrict__ p1, const float* __restrict__ q1,
    const float* __restrict__ pB, const float* __restrict__ pmp,
    const float* __restrict__ pmq, const float* __restrict__ vc,
    const float* __restrict__ vm, const int* __restrict__ qmask,
    _Float16* __restrict__ aggin) {
  const int bp = blockIdx.x;
  const int b  = bp >> 9;
  __shared__ float qbuf[NQ * NH2];                         // 64 KB
  __shared__ float p1row[NH], pBrow[NH2], pmprow[NH];
  __shared__ float s_c[NQ], s_b[NQ], s_m[NQ];
  __shared__ float a_c[NQ], a_b[NQ], a_m[NQ];
  const int t = threadIdx.x;
  for (int i = t; i < NQ * NH2; i += 256) qbuf[i] = query[(size_t)b * NQ * NH2 + i];
  if (t < NH) p1row[t]  = p1[(size_t)bp * NH + t];
  if (t < NH) pmprow[t] = pmp[(size_t)bp * NH + t];
  pBrow[t] = pB[(size_t)bp * NH2 + t];
  __syncthreads();
  if (t < 64) {
    const int q = t;
    const float* q1r = q1 + (size_t)(b * NQ + q) * NH;
    float s = 0.f;
    for (int h = 0; h < NH; ++h) s += vc[h] * tanhf(p1row[h] + q1r[h]);
    s_c[q] = qmask[b * NQ + q] ? s : NEG_INF;
  } else if (t < 128) {
    const int q = t - 64;
    float s = 0.f;
    for (int d = 0; d < NH2; ++d) s += pBrow[d] * qbuf[q * NH2 + d];
    s_b[q] = qmask[b * NQ + q] ? s : NEG_INF;
  } else if (t < 192) {
    const int q = t - 128;
    const float* pmqr = pmq + (size_t)(b * NQ + q) * NH;
    float s = 0.f;
    for (int h = 0; h < NH; ++h) s += vm[h] * tanhf(pmqr[h] - pmprow[h]);
    s_m[q] = qmask[b * NQ + q] ? s : NEG_INF;
  }
  __syncthreads();
  if (t < 3) {
    float* s = (t == 0) ? s_c : ((t == 1) ? s_b : s_m);
    float* a = (t == 0) ? a_c : ((t == 1) ? a_b : a_m);
    float m = NEG_INF;
    for (int i = 0; i < NQ; ++i) m = fmaxf(m, s[i]);
    float ss = 0.f;
    for (int i = 0; i < NQ; ++i) { float e = __expf(s[i] - m); a[i] = e; ss += e; }
    float inv = 1.f / ss;
    for (int i = 0; i < NQ; ++i) a[i] *= inv;
  }
  __syncthreads();
  float rc = 0.f, rb = 0.f, rm = 0.f;
  for (int q = 0; q < NQ; ++q) {
    const float qd = qbuf[q * NH2 + t];
    rc += a_c[q] * qd; rb += a_b[q] * qd; rm += a_m[q] * qd;
  }
  _Float16* row = aggin + (size_t)bp * NH10;
  row[t]        = (_Float16)passage[(size_t)bp * NH2 + t];  // cols 0..255
  row[256 + t]  = (_Float16)rc;                             // pc
  row[512 + t]  = (_Float16)rb;                             // pb
  row[1024 + t] = (_Float16)rm;                             // pm
}

// ---------------- fused attention d: WMMA GEMM + tanh·vd + softmax + attend --
__global__ void __launch_bounds__(256) k_attn_d(
    const float* __restrict__ passage, const float* __restrict__ query,
    const _Float16* __restrict__ WdP, const float* __restrict__ vd,
    const int* __restrict__ qmask, _Float16* __restrict__ aggin) {
  const int bp = blockIdx.x;
  const int b  = bp >> 9;
  __shared__ _Float16 qA[NQ * NH2];     // (query ∘ pass_row) in f16, 32 KB
  __shared__ float prow[NH2];
  __shared__ float score[NQ], alpha[NQ];
  const int t = threadIdx.x;
  prow[t] = passage[(size_t)bp * NH2 + t];
  if (t < NQ) score[t] = 0.f;
  __syncthreads();
  const float* qsrc = query + (size_t)b * NQ * NH2;
  for (int i = t; i < NQ * NH2; i += 256)
    qA[i] = (_Float16)(qsrc[i] * prow[i & (NH2 - 1)]);
  __syncthreads();

  const int lane = t & 31;
  const int wave = t >> 5;
  const int mt   = wave & 3;          // q-tile 0..3
  const int ng   = (wave >> 2) * 4;   // n-tile group base: 0 or 4
  v8f acc[4] = {};
  for (int k0 = 0; k0 < NH2; k0 += 32) {
    v16h a = load_frag_a(qA + mt * 16 * NH2 + k0, NH2, lane);
#pragma unroll
    for (int j = 0; j < 4; ++j) {
      v16h bf = load_frag_bp(WdP, k0 >> 5, ng + j, NH >> 4, lane);
      acc[j] = wmma32(a, bf, acc[j]);
    }
  }
  // reduce tanh(acc)*vd over the 128 h columns -> per-q partial score
#pragma unroll
  for (int v = 0; v < 8; ++v) {
    float s = 0.f;
#pragma unroll
    for (int j = 0; j < 4; ++j) {
      const int col = (ng + j) * 16 + (lane & 15);
      s += tanhf(acc[j][v]) * vd[col];
    }
    for (int off = 1; off < 16; off <<= 1) s += __shfl_xor(s, off, 32);
    if ((lane & 15) == 0) {
      const int q = mt * 16 + v + ((lane >= 16) ? 8 : 0);
      atomicAdd(&score[q], s);
    }
  }
  __syncthreads();
  if (t == 0) {
    float m = NEG_INF;
    for (int q = 0; q < NQ; ++q) {
      const float sv = qmask[b * NQ + q] ? score[q] : NEG_INF;
      score[q] = sv; m = fmaxf(m, sv);
    }
    float ss = 0.f;
    for (int q = 0; q < NQ; ++q) { float e = __expf(score[q] - m); alpha[q] = e; ss += e; }
    float inv = 1.f / ss;
    for (int q = 0; q < NQ; ++q) alpha[q] *= inv;
  }
  __syncthreads();
  float r = 0.f;
  const float* qb = qsrc + t;
  for (int q = 0; q < NQ; ++q) r += alpha[q] * qb[q * NH2];
  aggin[(size_t)bp * NH10 + 768 + t] = (_Float16)r;        // pd slice
}

// ---------------- bidirectional GRU scan (16 concurrent scans) ----------------
__global__ void __launch_bounds__(NG) k_gru(
    const float* __restrict__ gi_f, const float* __restrict__ gi_b,
    const float* __restrict__ Whh_f, const float* __restrict__ Whh_b,
    const float* __restrict__ bhh_f, const float* __restrict__ bhh_b,
    float* __restrict__ agg, _Float16* __restrict__ aggh) {
  const int b   = blockIdx.x & 7;
  const int dir = blockIdx.x >> 3;
  const float* gi  = dir ? gi_b  : gi_f;
  const float* Whh = dir ? Whh_b : Whh_f;
  const float* bhh = dir ? bhh_b : bhh_f;
  __shared__ float W[NH * NG];   // 192 KB of the 320 KB WGP LDS
  __shared__ float h[NH];
  __shared__ float gh[NG];
  const int t = threadIdx.x;
  for (int i = t; i < NH * NG; i += NG) W[i] = Whh[i];
  if (t < NH) h[t] = 0.f;
  const float bg = bhh[t];
  __syncthreads();
  for (int step = 0; step < NP; ++step) {
    const int tt = dir ? (NP - 1 - step) : step;
    const float* gir = gi + (size_t)(b * NP + tt) * NG;
    float s = bg;
    for (int k = 0; k < NH; ++k) s += h[k] * W[k * NG + t];
    gh[t] = s;
    __syncthreads();
    if (t < NH) {
      const float r  = 1.f / (1.f + __expf(-(gir[t]        + gh[t])));
      const float z  = 1.f / (1.f + __expf(-(gir[NH + t]   + gh[NH + t])));
      const float n  = tanhf(gir[2 * NH + t] + r * gh[2 * NH + t]);
      const float hn = (1.f - z) * n + z * h[t];
      h[t] = hn;
      const size_t o = (size_t)(b * NP + tt) * NH2 + dir * NH + t;
      agg[o] = hn; aggh[o] = (_Float16)hn;
    }
    __syncthreads();
  }
}

// ---------------- rq = softmax(tanh(q@wq)@vq) pooled query; rq2 = rq@wp2 ------
__global__ void __launch_bounds__(256) k_rq(
    const float* __restrict__ tq, const float* __restrict__ query,
    const float* __restrict__ vq, const float* __restrict__ wp2,
    const int* __restrict__ qmask, float* __restrict__ rq2out) {
  const int b = blockIdx.x;
  const int t = threadIdx.x;
  __shared__ float s[NQ], al[NQ], rq[NH2];
  if (t < NQ) {
    const float* r = tq + (size_t)(b * NQ + t) * NH;
    float v = 0.f;
    for (int h = 0; h < NH; ++h) v += tanhf(r[h]) * vq[h];
    s[t] = qmask[b * NQ + t] ? v : NEG_INF;
  }
  __syncthreads();
  if (t == 0) {
    float m = NEG_INF;
    for (int i = 0; i < NQ; ++i) m = fmaxf(m, s[i]);
    float ss = 0.f;
    for (int i = 0; i < NQ; ++i) { al[i] = __expf(s[i] - m); ss += al[i]; }
    float inv = 1.f / ss;
    for (int i = 0; i < NQ; ++i) al[i] *= inv;
  }
  __syncthreads();
  {
    float r = 0.f;
    const float* qb = query + (size_t)b * NQ * NH2 + t;
    for (int q = 0; q < NQ; ++q) r += al[q] * qb[q * NH2];
    rq[t] = r;
  }
  __syncthreads();
  if (t < NH) {
    float v = 0.f;
    for (int d = 0; d < NH2; ++d) v += rq[d] * wp2[d * NH + t];
    rq2out[b * NH + t] = v;
  }
}

// ---------------- sp softmax, rp pool, predict, logits ----------------
__global__ void __launch_bounds__(256) k_final(
    const float* __restrict__ aggW, const float* __restrict__ rq2,
    const float* __restrict__ vp, const float* __restrict__ agg,
    const float* __restrict__ predict_w, const float* __restrict__ a123,
    const int* __restrict__ pmask, float* __restrict__ out) {
  const int b = blockIdx.x;
  const int t = threadIdx.x;
  __shared__ float sp[NP], rp[NH2], rpw[NH], lg[3];
  const float* r2 = rq2 + b * NH;
  for (int p = t; p < NP; p += 256) {
    const float* ar = aggW + (size_t)(b * NP + p) * NH;
    float s = 0.f;
    for (int h = 0; h < NH; ++h) s += tanhf(ar[h] + r2[h]) * vp[h];
    sp[p] = pmask[b * NP + p] ? s : NEG_INF;
  }
  __syncthreads();
  if (t == 0) {
    float m = NEG_INF;
    for (int i = 0; i < NP; ++i) m = fmaxf(m, sp[i]);
    float ss = 0.f;
    for (int i = 0; i < NP; ++i) { float e = __expf(sp[i] - m); sp[i] = e; ss += e; }
    float inv = 1.f / ss;
    for (int i = 0; i < NP; ++i) sp[i] *= inv;
  }
  __syncthreads();
  {
    float v = 0.f;
    const float* ab = agg + (size_t)b * NP * NH2 + t;
    for (int p = 0; p < NP; ++p) v += sp[p] * ab[p * NH2];
    rp[t] = v;
  }
  __syncthreads();
  if (t < NH) {
    float v = 0.f;
    for (int d = 0; d < NH2; ++d) v += rp[d] * predict_w[d * NH + t];
    rpw[t] = v > 0.f ? v : 0.01f * v;                      // leaky_relu(0.01)
  }
  __syncthreads();
  if (t < 3) {
    const float* al_ = a123 + (b * 3 + t) * NH;
    float v = 0.f;
    for (int h = 0; h < NH; ++h) v += al_[h] * rpw[h];
    lg[t] = v;
  }
  __syncthreads();
  if (t == 0) {
    const float m = fmaxf(lg[0], fmaxf(lg[1], lg[2]));
    const float e0 = __expf(lg[0] - m), e1 = __expf(lg[1] - m), e2 = __expf(lg[2] - m);
    const float ss = e0 + e1 + e2;
    out[b * 3 + 0] = e0 / ss; out[b * 3 + 1] = e1 / ss; out[b * 3 + 2] = e2 / ss;
  }
}

// ======================== host-side orchestration ========================
extern "C" void kernel_launch(void* const* d_in, const int* in_sizes, int n_in,
                              void* d_out, int out_size, void* d_ws, size_t ws_size,
                              hipStream_t stream) {
  (void)in_sizes; (void)n_in; (void)out_size; (void)ws_size;
  const float* passage   = (const float*)d_in[0];
  const float* query     = (const float*)d_in[1];
  const float* alt1      = (const float*)d_in[2];
  const float* alt2      = (const float*)d_in[3];
  const float* alt3      = (const float*)d_in[4];
  const float* a_att_w   = (const float*)d_in[5];
  const float* Wc1       = (const float*)d_in[6];
  const float* Wc2       = (const float*)d_in[7];
  const float* vc        = (const float*)d_in[8];
  const float* Wb        = (const float*)d_in[9];
  const float* Wd        = (const float*)d_in[10];
  const float* vd        = (const float*)d_in[11];
  const float* Wm        = (const float*)d_in[12];
  const float* vm        = (const float*)d_in[13];
  const float* wq        = (const float*)d_in[14];
  const float* vq        = (const float*)d_in[15];
  const float* wp1       = (const float*)d_in[16];
  const float* wp2       = (const float*)d_in[17];
  const float* vp        = (const float*)d_in[18];
  const float* predict_w = (const float*)d_in[19];
  const float* gWih_f    = (const float*)d_in[20];
  const float* gWhh_f    = (const float*)d_in[21];
  const float* gbih_f    = (const float*)d_in[22];
  const float* gbhh_f    = (const float*)d_in[23];
  const float* gWih_b    = (const float*)d_in[24];
  const float* gWhh_b    = (const float*)d_in[25];
  const float* gbih_b    = (const float*)d_in[26];
  const float* gbhh_b    = (const float*)d_in[27];
  const int* pmask  = (const int*)d_in[28];
  const int* qmask  = (const int*)d_in[29];
  const int* a1mask = (const int*)d_in[30];
  const int* a2mask = (const int*)d_in[31];
  const int* a3mask = (const int*)d_in[32];
  float* out = (float*)d_out;

  char* ws = (char*)d_ws;
  size_t off = 0;
  auto alloc = [&](size_t bytes) -> void* {
    off = (off + 255) & ~(size_t)255;
    void* p = ws + off;
    off += bytes;
    return p;
  };
  const int MP = NB * NP;   // 4096
  const int MQ = NB * NQ;   // 512

  _Float16* passage_h = (_Float16*)alloc((size_t)MP * NH2 * 2);
  _Float16* query_h   = (_Float16*)alloc((size_t)MQ * NH2 * 2);
  _Float16* Wc1_p  = (_Float16*)alloc(NH2 * NH * 2);
  _Float16* Wc2_p  = (_Float16*)alloc(NH2 * NH * 2);
  _Float16* Wb_p   = (_Float16*)alloc(NH2 * NH2 * 2);
  _Float16* Wd_p   = (_Float16*)alloc(NH2 * NH * 2);
  _Float16* Wm_p   = (_Float16*)alloc(NH2 * NH * 2);
  _Float16* wq_p   = (_Float16*)alloc(NH2 * NH * 2);
  _Float16* wp1_p  = (_Float16*)alloc(NH2 * NH * 2);
  _Float16* gWihf_p = (_Float16*)alloc((size_t)NH10 * NG * 2);
  _Float16* gWihb_p = (_Float16*)alloc((size_t)NH10 * NG * 2);
  _Float16* aggin_h = (_Float16*)alloc((size_t)MP * NH10 * 2);
  _Float16* agg_h   = (_Float16*)alloc((size_t)MP * NH2 * 2);
  float* p1   = (float*)alloc((size_t)MP * NH * 4);
  float* q1   = (float*)alloc((size_t)MQ * NH * 4);
  float* pB   = (float*)alloc((size_t)MP * NH2 * 4);
  float* pmp  = (float*)alloc((size_t)MP * NH * 4);
  float* pmq  = (float*)alloc((size_t)MQ * NH * 4);
  float* tq   = (float*)alloc((size_t)MQ * NH * 4);
  float* gi_f = (float*)alloc((size_t)MP * NG * 4);
  float* gi_b = (float*)alloc((size_t)MP * NG * 4);
  float* agg  = (float*)alloc((size_t)MP * NH2 * 4);
  float* aggW = (float*)alloc((size_t)MP * NH * 4);
  float* a123 = (float*)alloc(NB * 3 * NH * 4);
  float* rq2  = (float*)alloc(NB * NH * 4);

  auto cvt = [&](const float* s, _Float16* d, int n) {
    k_cvt<<<(n + 255) / 256, 256, 0, stream>>>(s, d, n);
  };
  auto pack = [&](const float* s, _Float16* d, int K, int N) {
    int n = K * N;
    k_pack_b<<<(n + 255) / 256, 256, 0, stream>>>(s, d, K, N);
  };
  auto gemm = [&](const _Float16* A, const _Float16* Bp, const float* bias,
                  float* C, int M, int N, int K) {
    dim3 g(N / 64, M / 64);
    k_wmma_gemm<<<g, 128, 0, stream>>>(A, Bp, bias, C, M, N, K);
  };

  // ---- stage 0: f16 conversions (A operands) + fragment-packing (B operands)
  cvt(passage, passage_h, MP * NH2);
  cvt(query,   query_h,   MQ * NH2);
  pack(Wc1, Wc1_p, NH2, NH);
  pack(Wc2, Wc2_p, NH2, NH);
  pack(Wb,  Wb_p,  NH2, NH2);
  pack(Wd,  Wd_p,  NH2, NH);
  pack(Wm,  Wm_p,  NH2, NH);
  pack(wq,  wq_p,  NH2, NH);
  pack(wp1, wp1_p, NH2, NH);
  pack(gWih_f, gWihf_p, NH10, NG);
  pack(gWih_b, gWihb_p, NH10, NG);

  // ---- stage 1: independent projections (WMMA)
  gemm(passage_h, Wc1_p, nullptr, p1,  MP, NH,  NH2);
  gemm(query_h,   Wc2_p, nullptr, q1,  MQ, NH,  NH2);
  gemm(passage_h, Wb_p,  nullptr, pB,  MP, NH2, NH2);
  gemm(passage_h, Wm_p,  nullptr, pmp, MP, NH,  NH2);
  gemm(query_h,   Wm_p,  nullptr, pmq, MQ, NH,  NH2);
  gemm(query_h,   wq_p,  nullptr, tq,  MQ, NH,  NH2);

  // ---- stage 2: alternative pooling
  k_alt_pool<<<NB, 128, 0, stream>>>(alt1, a1mask, a_att_w, a123, 0);
  k_alt_pool<<<NB, 128, 0, stream>>>(alt2, a2mask, a_att_w, a123, 1);
  k_alt_pool<<<NB, 128, 0, stream>>>(alt3, a3mask, a_att_w, a123, 2);

  // ---- stage 3: fused attentions -> agg_in (f16)
  k_attn_cbm<<<MP, 256, 0, stream>>>(passage, query, p1, q1, pB, pmp, pmq,
                                     vc, vm, qmask, aggin_h);
  k_attn_d<<<MP, 256, 0, stream>>>(passage, query, Wd_p, vd, qmask, aggin_h);

  // ---- stage 4: GRU input gates (WMMA), then 16-way parallel scan
  gemm(aggin_h, gWihf_p, gbih_f, gi_f, MP, NG, NH10);
  gemm(aggin_h, gWihb_p, gbih_b, gi_b, MP, NG, NH10);
  k_gru<<<2 * NB, NG, 0, stream>>>(gi_f, gi_b, gWhh_f, gWhh_b, gbhh_f, gbhh_b,
                                   agg, agg_h);

  // ---- stage 5: pointer epilogue
  gemm(agg_h, wp1_p, nullptr, aggW, MP, NH, NH2);
  k_rq<<<NB, 256, 0, stream>>>(tq, query, vq, wp2, qmask, rq2);
  k_final<<<NB, 256, 0, stream>>>(aggW, rq2, vp, agg, predict_w, a123, pmask, out);
}
